// CrossAttention_39350490366723
// MI455X (gfx1250) — compile-verified, matrix operand order checked by reading, not run on hardware
//
#include <hip/hip_runtime.h>
#include <hip/hip_bf16.h>

typedef __attribute__((ext_vector_type(16))) __bf16 v16bf;
typedef __attribute__((ext_vector_type(8)))  float  v8f;
typedef __attribute__((ext_vector_type(4)))  unsigned int u32x4;
typedef __attribute__((ext_vector_type(8)))  int i32x8;
typedef __attribute__((ext_vector_type(4)))  int i32x4;

#define DIM   512
#define LDIM  256
#define NHEAD 8
#define HD    64
#define BT    32
#define TT    1024
#define NTOK  77
#define NP    80      // S padding (5 * 16 col tiles)
#define NP2   96      // P/V K-dim padding (3 * 32)

__device__ __forceinline__ v16bf frag_from(uint4 lo, uint4 hi) {
  union { uint4 u[2]; v16bf v; } x;
  x.u[0] = lo; x.u[1] = hi;
  return x.v;
}

__device__ __forceinline__ v8f bwmma(v16bf a, v16bf b, v8f c) {
  // D = A(16x32 bf16) * B(32x16 bf16) + C(16x16 f32)
  return __builtin_amdgcn_wmma_f32_16x16x32_bf16(
      /*neg_a=*/false, a, /*neg_b=*/false, b,
      /*c_mod=*/(short)0, c, /*reuse_a=*/false, /*reuse_b=*/false);
}

// ---------------------------------------------------------------------------
// Tensor Data Mover: async 2D bf16 tile copy global -> LDS.
//   tile = rows x width (bf16 elements), global row stride = stride elems.
//   tensor_dim1 = avail_rows: rows beyond it are zero-filled by TDM OOB logic.
// Issue from ONE wave only; fence with s_wait_tensorcnt before consuming.
// ---------------------------------------------------------------------------
__device__ __forceinline__ void tdm_load_2d_bf16(unsigned int lds_addr, const void* gptr,
                                                 unsigned int width, unsigned int rows,
                                                 unsigned int avail_rows, unsigned int stride) {
  const unsigned long long ga = (unsigned long long)(uintptr_t)gptr;
  u32x4 g0;
  g0[0] = 1u;                                   // count=1 (valid user descriptor)
  g0[1] = lds_addr;                             // LDS byte address [63:32]
  g0[2] = (unsigned int)ga;                     // global_addr[31:0]
  g0[3] = (unsigned int)((ga >> 32) & 0x1FFFFFFu) | (2u << 30);  // addr[56:32] | type=2
  i32x8 g1;
  g1[0] = (int)(1u << 16);                      // data_size=1 (2 bytes/elem)
  g1[1] = (int)(width << 16);                   // tensor_dim0[15:0] @ [63:48]
  g1[2] = (int)((width >> 16) | (avail_rows << 16));   // tdim0 hi | tensor_dim1 lo
  g1[3] = (int)((avail_rows >> 16) | (width << 16));   // tdim1 hi | tile_dim0
  g1[4] = (int)rows;                            // tile_dim1 (tile_dim2=0)
  g1[5] = (int)stride;                          // tensor_dim0_stride[31:0]
  g1[6] = 0;                                    // stride hi | tensor_dim1_stride lo
  g1[7] = 0;
  const i32x4 z4 = {0, 0, 0, 0};                // groups 2/3 unused (2D tensor)
#if __clang_major__ >= 23
  const i32x8 z8 = {0, 0, 0, 0, 0, 0, 0, 0};
  __builtin_amdgcn_tensor_load_to_lds(g0, g1, z4, z4, z8, 0);
#else
  __builtin_amdgcn_tensor_load_to_lds(g0, g1, z4, z4, 0);
#endif
}

// ---------------------------------------------------------------------------
// LayerNorm: one wave32 per row, fp32 in -> bf16 out.
// ---------------------------------------------------------------------------
__global__ void ln_bf16_kernel(const float* __restrict__ in, __bf16* __restrict__ out,
                               const float* __restrict__ gamma, const float* __restrict__ beta,
                               int nrows, int R) {
  const int lane = threadIdx.x & 31;
  const int row  = blockIdx.x * (blockDim.x >> 5) + (threadIdx.x >> 5);
  if (row >= nrows) return;
  const float* __restrict__ p = in + (size_t)row * R;

  float s = 0.f, s2 = 0.f;
  for (int i = lane; i < R; i += 32) {
    float t = p[i];
    s += t; s2 += t * t;
  }
#pragma unroll
  for (int o = 16; o; o >>= 1) {
    s  += __shfl_xor(s,  o, 32);
    s2 += __shfl_xor(s2, o, 32);
  }
  const float m  = s / (float)R;
  const float vr = s2 / (float)R - m * m;
  const float rs = rsqrtf(vr + 1e-5f);

  __bf16* __restrict__ q = out + (size_t)row * R;
  for (int i = lane; i < R; i += 32)
    q[i] = (__bf16)((p[i] - m) * rs * gamma[i] + beta[i]);
}

// ---------------------------------------------------------------------------
// C[M x 512] = A[M x Kd](bf16,row-major) * W[Kd x 512](fp32) + bias, bf16 out.
// Block: 128 threads (4 waves), 64x64 output tile. grid = (ceil(M/64), 8).
// A tile staged by TDM (async, zero-fills partial M tiles); W tile converted
// fp32->bf16 and transposed by the block while the TDM runs.
// ---------------------------------------------------------------------------
__global__ void wmma_gemm_bias_kernel(const __bf16* __restrict__ A,
                                      const float* __restrict__ W,
                                      const float* __restrict__ bias,
                                      __bf16* __restrict__ C,
                                      int M, int Kd) {
  __shared__ __align__(16) __bf16 ldsA[64 * 32];   // [row][k]
  __shared__ __align__(16) __bf16 ldsBt[64 * 32];  // [col][k] (transposed W tile)

  const int t    = threadIdx.x;
  const int wave = t >> 5;
  const int lane = t & 31;
  const int g    = lane >> 4;
  const int ln   = lane & 15;
  const int rowBase = blockIdx.x * 64;
  const int c0      = blockIdx.y * 64;
  const unsigned int availRows = (unsigned int)(M - rowBase);

  v8f acc[4] = {v8f{}, v8f{}, v8f{}, v8f{}};

  for (int kk = 0; kk < Kd; kk += 32) {
    // Async stage of A tile 64x32 bf16 via Tensor Data Mover (wave 0 issues).
    if (wave == 0) {
      tdm_load_2d_bf16((unsigned int)(uintptr_t)&ldsA[0],
                       A + (size_t)rowBase * Kd + kk,
                       /*width=*/32, /*rows=*/64, availRows, (unsigned int)Kd);
    }
    // Meanwhile: stage W tile 32x64, fp32 -> bf16, transposed into [col][k].
    // Pack bf16 pairs along K so every LDS store is a b32.
    {
      const int col = t & 63;
      const int kp0 = (t >> 6) * 8;     // 8 k-pairs per thread
#pragma unroll
      for (int kp = kp0; kp < kp0 + 8; ++kp) {
        const float w0 = W[(size_t)(kk + 2 * kp)     * DIM + c0 + col];
        const float w1 = W[(size_t)(kk + 2 * kp + 1) * DIM + c0 + col];
        union { __bf16 h[2]; unsigned int u; } pk;
        pk.h[0] = (__bf16)w0; pk.h[1] = (__bf16)w1;
        *(unsigned int*)&ldsBt[col * 32 + 2 * kp] = pk.u;
      }
    }
    if (wave == 0) __builtin_amdgcn_s_wait_tensorcnt(0);
    __syncthreads();

    // A fragment: 16 rows of this wave, K=0..31 of staged tile.
    const __bf16* ar = &ldsA[(wave * 16 + ln) * 32 + g * 8];
    v16bf afrag = frag_from(*(const uint4*)ar, *(const uint4*)(ar + 16));
#pragma unroll
    for (int nt = 0; nt < 4; ++nt) {
      const __bf16* br = &ldsBt[(nt * 16 + ln) * 32 + g * 8];
      v16bf bfrag = frag_from(*(const uint4*)br, *(const uint4*)(br + 16));
      acc[nt] = bwmma(afrag, bfrag, acc[nt]);
    }
    __syncthreads();
  }

  // Epilogue: +bias, convert bf16, scatter-store.
#pragma unroll
  for (int nt = 0; nt < 4; ++nt) {
    const int col = c0 + nt * 16 + ln;
    const float bv = bias[col];
#pragma unroll
    for (int j = 0; j < 8; ++j) {
      const int row = rowBase + wave * 16 + j + 8 * g;
      if (row < M)
        C[(size_t)row * DIM + col] = (__bf16)(acc[nt][j] + bv);
    }
  }
}

// ---------------------------------------------------------------------------
// Attention: one block per (b, h, 64 query rows). 128 threads = 4 waves,
// each wave owns a 16-row Q tile. K staged by TDM; V staged transposed.
// grid = (T/64, B*H).
// ---------------------------------------------------------------------------
__global__ void attn_kernel(const __bf16* __restrict__ Q,
                            const __bf16* __restrict__ Kb,
                            const __bf16* __restrict__ Vb,
                            float* __restrict__ out) {
  __shared__ __align__(16) __bf16 ldsK [NP * HD];      // [tok][d], tok padded to 80
  __shared__ __align__(16) __bf16 ldsVt[HD * NP2];     // [d][tok], tok padded to 96
  __shared__ __align__(16) float  ldsS [4][16 * NP];   // per-wave scores
  __shared__ __align__(16) __bf16 ldsP [4][16 * NP2];  // per-wave probabilities

  const int t    = threadIdx.x;
  const int wave = t >> 5;
  const int lane = t & 31;
  const int g    = lane >> 4;
  const int ln   = lane & 15;
  const int b    = blockIdx.y >> 3;
  const int h    = blockIdx.y & 7;
  const int t0   = blockIdx.x * 64;

  // Async stage of K tile (77 x 64 bf16, row stride 512) via TDM.
  if (wave == 0) {
    tdm_load_2d_bf16((unsigned int)(uintptr_t)&ldsK[0],
                     Kb + (size_t)(b * NTOK) * DIM + h * HD,
                     /*width=*/HD, /*rows=*/NTOK, /*avail=*/NTOK, /*stride=*/DIM);
  }
  // Zero the pad rows 77..79 of ldsK (disjoint from TDM destination).
  for (int i = t; i < (NP - NTOK) * HD; i += 128)
    ldsK[NTOK * HD + i] = (__bf16)0.f;
  // Stage V transposed (zero-padded) : ldsVt[d][n]
  for (int i = t; i < HD * NP2; i += 128) {
    const int d = i / NP2, n = i % NP2;
    ldsVt[i] = (n < NTOK) ? Vb[(size_t)(b * NTOK + n) * DIM + h * HD + d] : (__bf16)0.f;
  }
  if (wave == 0) __builtin_amdgcn_s_wait_tensorcnt(0);
  __syncthreads();

  const int tq = t0 + wave * 16;

  // Q fragments (two K=32 halves of the 16x64 tile), straight from global.
  const __bf16* qr = Q + (size_t)(b * TT + tq + ln) * DIM + h * HD + g * 8;
  v16bf a0 = frag_from(*(const uint4*)qr,        *(const uint4*)(qr + 16));
  v16bf a1 = frag_from(*(const uint4*)(qr + 32), *(const uint4*)(qr + 48));

  const float scale = 0.125f;  // 1/sqrt(64)

  // S = Q * K^T : 5 column tiles x 2 WMMAs.
#pragma unroll
  for (int ct = 0; ct < 5; ++ct) {
    v8f acc = v8f{};
    const __bf16* kr = &ldsK[(ct * 16 + ln) * HD + g * 8];
    v16bf b0 = frag_from(*(const uint4*)kr,        *(const uint4*)(kr + 16));
    v16bf b1 = frag_from(*(const uint4*)(kr + 32), *(const uint4*)(kr + 48));
    acc = bwmma(a0, b0, acc);
    acc = bwmma(a1, b1, acc);
#pragma unroll
    for (int j = 0; j < 8; ++j)
      ldsS[wave][(j + 8 * g) * NP + ct * 16 + ln] = acc[j] * scale;
  }
  __syncthreads();

  // Softmax over the 77 real key tokens; lanes 0..15 own one row each.
  if (lane < 16) {
    float* srow = &ldsS[wave][lane * NP];
    float mx = -1e30f;
    for (int n = 0; n < NTOK; ++n) mx = fmaxf(mx, srow[n]);
    float sum = 0.f;
    for (int n = 0; n < NTOK; ++n) {
      const float e = __expf(srow[n] - mx);
      srow[n] = e; sum += e;
    }
    const float inv = 1.f / sum;
    __bf16* prow = &ldsP[wave][lane * NP2];
    for (int n = 0; n < NTOK; ++n) prow[n] = (__bf16)(srow[n] * inv);
    for (int n = NTOK; n < NP2; ++n) prow[n] = (__bf16)0.f;
  }
  __syncthreads();

  // O = P * V : P is 16x96 (3 K=32 fragments), output 16x64 (4 col tiles).
  v16bf p[3];
#pragma unroll
  for (int kb = 0; kb < 3; ++kb) {
    const __bf16* pr = &ldsP[wave][ln * NP2 + kb * 32 + g * 8];
    p[kb] = frag_from(*(const uint4*)pr, *(const uint4*)(pr + 16));
  }
#pragma unroll
  for (int dt = 0; dt < 4; ++dt) {
    v8f acc = v8f{};
#pragma unroll
    for (int kb = 0; kb < 3; ++kb) {
      const __bf16* vr = &ldsVt[(dt * 16 + ln) * NP2 + kb * 32 + g * 8];
      v16bf bf = frag_from(*(const uint4*)vr, *(const uint4*)(vr + 16));
      acc = bwmma(p[kb], bf, acc);
    }
#pragma unroll
    for (int j = 0; j < 8; ++j)
      out[(size_t)(b * TT + tq + j + 8 * g) * DIM + h * HD + dt * 16 + ln] = acc[j];
  }
}

// ---------------------------------------------------------------------------
// Host-side launch.
// Inputs (all fp32): x, xf, ln_g, ln_b, tln_g, tln_b, Wq, bq, Wk, bk, Wv, bv
// ---------------------------------------------------------------------------
extern "C" void kernel_launch(void* const* d_in, const int* in_sizes, int n_in,
                              void* d_out, int out_size, void* d_ws, size_t ws_size,
                              hipStream_t stream) {
  const float* x     = (const float*)d_in[0];
  const float* xf    = (const float*)d_in[1];
  const float* ln_g  = (const float*)d_in[2];
  const float* ln_b  = (const float*)d_in[3];
  const float* tln_g = (const float*)d_in[4];
  const float* tln_b = (const float*)d_in[5];
  const float* Wq    = (const float*)d_in[6];
  const float* bq    = (const float*)d_in[7];
  const float* Wk    = (const float*)d_in[8];
  const float* bk    = (const float*)d_in[9];
  const float* Wv    = (const float*)d_in[10];
  const float* bv    = (const float*)d_in[11];
  float* out = (float*)d_out;

  const int MQ = BT * TT;     // 32768 query rows
  const int MK = BT * NTOK;   // 2464 key/value rows

  // Workspace layout (bytes): xn | xfn | Q | K | V, all bf16 (~70 MB).
  char* ws = (char*)d_ws;
  __bf16* xn  = (__bf16*)(ws);
  __bf16* xfn = (__bf16*)(ws + (size_t)MQ * DIM * 2);
  __bf16* Qb  = (__bf16*)(ws + (size_t)MQ * DIM * 2 + (size_t)MK * LDIM * 2);
  __bf16* Kb  = (__bf16*)((char*)Qb + (size_t)MQ * DIM * 2);
  __bf16* Vb  = (__bf16*)((char*)Kb + (size_t)MK * DIM * 2);

  // 1) LayerNorms (8 rows per 256-thread block).
  ln_bf16_kernel<<<MQ / 8, 256, 0, stream>>>(x, xn, ln_g, ln_b, MQ, DIM);
  ln_bf16_kernel<<<(MK + 7) / 8, 256, 0, stream>>>(xf, xfn, tln_g, tln_b, MK, LDIM);

  // 2) Projections via WMMA GEMM (TDM-staged A tiles).
  wmma_gemm_bias_kernel<<<dim3(MQ / 64, DIM / 64), 128, 0, stream>>>(xn, Wq, bq, Qb, MQ, DIM);
  wmma_gemm_bias_kernel<<<dim3((MK + 63) / 64, DIM / 64), 128, 0, stream>>>(xfn, Wk, bk, Kb, MK, LDIM);
  wmma_gemm_bias_kernel<<<dim3((MK + 63) / 64, DIM / 64), 128, 0, stream>>>(xfn, Wv, bv, Vb, MK, LDIM);

  // 3) Attention (softmax over N=77) writing fp32 output.
  attn_kernel<<<dim3(TT / 64, BT * NHEAD), 128, 0, stream>>>(Qb, Kb, Vb, out);
}
